// GNN_18176301596804
// MI455X (gfx1250) — matrix operand order repaired and checked by reading, not compiled, not run on hardware
//
#include <hip/hip_runtime.h>
#include <hip/hip_bf16.h>

#define NN 100000
#define DD 128
#define EE 600000
#define LL 5
#define GN_EPS 1e-5f
#define NTILES (NN / 16)     // 6250
#define BSTRIDE 136          // padded LDS row stride (bf16 elems): bank base = 4n mod 64

typedef __attribute__((ext_vector_type(16))) __bf16 v16bf;
typedef __attribute__((ext_vector_type(8)))  __bf16 v8bf;
typedef __attribute__((ext_vector_type(8)))  float  v8f;

// =============== CSR build (once per call; edges are layer-invariant) ===============
__global__ void k_zero_deg(unsigned* __restrict__ deg) {
    int i = blockIdx.x * blockDim.x + threadIdx.x;
    if (i < NN) deg[i] = 0u;
}

__global__ void k_hist(const int* __restrict__ ei, unsigned* __restrict__ deg) {
    int e = blockIdx.x * blockDim.x + threadIdx.x;
    if (e < EE) atomicAdd(&deg[ei[EE + e]], 1u);
}

// single-workgroup chunked Hillis-Steele exclusive scan over deg[0..N)
__global__ void k_scan(const unsigned* __restrict__ deg, unsigned* __restrict__ off,
                       unsigned* __restrict__ cursor) {
    __shared__ unsigned buf[1024];
    __shared__ unsigned carry;
    int tid = threadIdx.x;
    if (tid == 0) carry = 0u;
    __syncthreads();
    for (int base = 0; base < NN; base += 1024) {
        int i = base + tid;
        unsigned v = (i < NN) ? deg[i] : 0u;
        buf[tid] = v;
        __syncthreads();
        for (int s = 1; s < 1024; s <<= 1) {
            unsigned t = (tid >= s) ? buf[tid - s] : 0u;
            __syncthreads();
            buf[tid] += t;
            __syncthreads();
        }
        unsigned excl = carry + buf[tid] - v;
        if (i < NN) { off[i] = excl; cursor[i] = excl; }
        __syncthreads();                  // everyone read carry before update
        if (tid == 0) carry += buf[1023];
        __syncthreads();
    }
}

__global__ void k_fill(const int* __restrict__ ei, unsigned* __restrict__ cursor,
                       int* __restrict__ srcs) {
    int e = blockIdx.x * blockDim.x + threadIdx.x;
    if (e < EE) {
        int d = ei[EE + e];
        unsigned p = atomicAdd(&cursor[d], 1u);
        srcs[p] = ei[e];
    }
}

// =============== aggregation: acc[i] = h[i] + sum_{j->i} h[j] (gather, no fp atomics) ===============
__global__ void k_aggregate(const float* __restrict__ h, const unsigned* __restrict__ off,
                            const unsigned* __restrict__ deg, const int* __restrict__ srcs,
                            float* __restrict__ acc) {
    int node = blockIdx.x * 8 + (threadIdx.x >> 5);   // one wave per node, exact grid
    int lane = threadIdx.x & 31;
    const float4* hp = (const float4*)h;
    float4 a = hp[(size_t)node * 32 + lane];
    unsigned s0 = off[node];
    unsigned s1 = s0 + deg[node];
    for (unsigned j = s0; j < s1; ++j) {
        int s = srcs[j];                              // wave-uniform
        float4 v = hp[(size_t)s * 32 + lane];
        a.x += v.x; a.y += v.y; a.z += v.z; a.w += v.w;
    }
    ((float4*)acc)[(size_t)node * 32 + lane] = a;
}

// =============== misc ===============
__global__ void k_zero_stats(float* __restrict__ stats) { stats[threadIdx.x] = 0.0f; }

__global__ void k_convert(const float* __restrict__ W1, const float* __restrict__ W2,
                          __bf16* __restrict__ w1t, __bf16* __restrict__ w2t) {
    int idx = blockIdx.x * blockDim.x + threadIdx.x;   // < L*D*D, exact grid
    int l = idx / (DD * DD);
    int r = idx % (DD * DD);
    int k = r / DD;
    int n = r % DD;
    size_t o = (size_t)l * DD * DD + (size_t)n * DD + k;
    w1t[o] = (__bf16)W1[idx];
    w2t[o] = (__bf16)W2[idx];
}

// =============== shared helpers for the two WMMA GEMMs ===============
__device__ __forceinline__ void stage_B_to_lds(const __bf16* __restrict__ Bt, __bf16* Bs) {
    int t  = threadIdx.x;
    int r  = t >> 1;                 // 0..127
    int c0 = (t & 1) * 64;           // 0 or 64
    const uint4* src = (const uint4*)(Bt + (size_t)r * DD + c0);
    uint4* dst = (uint4*)(&Bs[r * BSTRIDE + c0]);
    #pragma unroll
    for (int i = 0; i < 8; ++i) dst[i] = src[i];
}

__device__ __forceinline__ v16bf load_bfrag(const __bf16* Bs, int n, int kb2) {
    // 16-byte aligned LDS reads, conflict-free (bank base 4n mod 64)
    v8bf lo = *(const v8bf*)(&Bs[n * BSTRIDE + kb2]);
    v8bf hi = *(const v8bf*)(&Bs[n * BSTRIDE + kb2 + 8]);
    return __builtin_shufflevector(lo, hi, 0, 1, 2, 3, 4, 5, 6, 7,
                                   8, 9, 10, 11, 12, 13, 14, 15);
}

// =============== GEMM1: Out = A @ W1 + b1 (in-place safe) + column moments ===============
__global__ void __launch_bounds__(256) k_gemm1(const float* __restrict__ A,
                                               const __bf16* __restrict__ Bt,
                                               const float* __restrict__ bias,
                                               float* __restrict__ Out,
                                               float* __restrict__ colsum,
                                               float* __restrict__ colsumsq) {
    __shared__ __bf16 Bs[DD * BSTRIDE];
    stage_B_to_lds(Bt, Bs);
    __syncthreads();

    int tile = blockIdx.x * 8 + (threadIdx.x >> 5);
    if (tile >= NTILES) return;                  // wave-uniform, after barrier
    int lane = threadIdx.x & 31;
    int m    = lane & 15;
    int half = lane >> 4;
    int row0 = tile * 16;
    const float* arow = A + (size_t)(row0 + m) * DD;

    v8f acc[8];
    #pragma unroll
    for (int nt = 0; nt < 8; ++nt) {
        float bv = bias[nt * 16 + m];
        #pragma unroll
        for (int j = 0; j < 8; ++j) acc[nt][j] = bv;
    }

    // register double-buffered B fragments: ds_load of frag i+1 overlaps wmma i
    v16bf bcur = load_bfrag(Bs, m, half * 16);
    #pragma unroll
    for (int ks = 0; ks < 4; ++ks) {
        int kb = ks * 32 + half * 16;
        v16bf af;
        #pragma unroll
        for (int i = 0; i < 8; ++i) {
            af[i]     = (__bf16)arow[kb + i];
            af[8 + i] = (__bf16)arow[kb + 8 + i];
        }
        #pragma unroll
        for (int nt = 0; nt < 8; ++nt) {
            int nxt = ks * 8 + nt + 1;          // compile-time after unroll
            v16bf bnext = bcur;
            if (nxt < 32) {
                int ks2 = nxt >> 3, nt2 = nxt & 7;
                bnext = load_bfrag(Bs, nt2 * 16 + m, ks2 * 32 + half * 16);
            }
            acc[nt] = __builtin_amdgcn_wmma_f32_16x16x32_bf16(
                false, af, false, bcur, (short)0, acc[nt], false, false);
            bcur = bnext;
        }
    }

    #pragma unroll
    for (int nt = 0; nt < 8; ++nt) {
        int n = nt * 16 + m;
        float s = 0.0f, s2 = 0.0f;
        #pragma unroll
        for (int v = 0; v < 8; ++v) {
            float y = acc[nt][v];
            s  += y;
            s2 += y * y;
            Out[(size_t)(row0 + half * 8 + v) * DD + n] = y;
        }
        unsafeAtomicAdd(&colsum[n],   s);
        unsafeAtomicAdd(&colsumsq[n], s2);
    }
}

// =============== GraphNorm finalize ===============
__global__ void k_finalize(const float* __restrict__ colsum, const float* __restrict__ colsumsq,
                           const float* __restrict__ gw, const float* __restrict__ gb,
                           const float* __restrict__ ga,
                           float* __restrict__ scale, float* __restrict__ shift) {
    int n = threadIdx.x;
    const float invN = 1.0f / (float)NN;
    float mean = colsum[n]   * invN;
    float ex2  = colsumsq[n] * invN;
    float a    = ga[n];
    float var  = ex2 - 2.0f * a * mean * mean + a * a * mean * mean;
    float rs   = rsqrtf(var + GN_EPS);
    float sc   = gw[n] * rs;
    scale[n] = sc;
    shift[n] = gb[n] - sc * a * mean;
}

// =============== GEMM2: Out = relu( relu(scale*x+shift) @ W2 + b2 ) ===============
__global__ void __launch_bounds__(256) k_gemm2(const float* __restrict__ A,
                                               const __bf16* __restrict__ Bt,
                                               const float* __restrict__ bias,
                                               const float* __restrict__ scale,
                                               const float* __restrict__ shift,
                                               float* __restrict__ Out) {
    __shared__ __bf16 Bs[DD * BSTRIDE];
    stage_B_to_lds(Bt, Bs);
    __syncthreads();

    int tile = blockIdx.x * 8 + (threadIdx.x >> 5);
    if (tile >= NTILES) return;
    int lane = threadIdx.x & 31;
    int m    = lane & 15;
    int half = lane >> 4;
    int row0 = tile * 16;
    const float* arow = A + (size_t)(row0 + m) * DD;

    v8f acc[8];
    #pragma unroll
    for (int nt = 0; nt < 8; ++nt) {
        float bv = bias[nt * 16 + m];
        #pragma unroll
        for (int j = 0; j < 8; ++j) acc[nt][j] = bv;
    }

    v16bf bcur = load_bfrag(Bs, m, half * 16);
    #pragma unroll
    for (int ks = 0; ks < 4; ++ks) {
        int kb = ks * 32 + half * 16;
        v16bf af;
        #pragma unroll
        for (int i = 0; i < 8; ++i) {
            int k0 = kb + i;
            int k1 = kb + 8 + i;
            float t0 = fmaxf(fmaf(scale[k0], arow[k0], shift[k0]), 0.0f);
            float t1 = fmaxf(fmaf(scale[k1], arow[k1], shift[k1]), 0.0f);
            af[i]     = (__bf16)t0;
            af[8 + i] = (__bf16)t1;
        }
        #pragma unroll
        for (int nt = 0; nt < 8; ++nt) {
            int nxt = ks * 8 + nt + 1;
            v16bf bnext = bcur;
            if (nxt < 32) {
                int ks2 = nxt >> 3, nt2 = nxt & 7;
                bnext = load_bfrag(Bs, nt2 * 16 + m, ks2 * 32 + half * 16);
            }
            acc[nt] = __builtin_amdgcn_wmma_f32_16x16x32_bf16(
                false, af, false, bcur, (short)0, acc[nt], false, false);
            bcur = bnext;
        }
    }

    #pragma unroll
    for (int nt = 0; nt < 8; ++nt) {
        int n = nt * 16 + m;
        #pragma unroll
        for (int v = 0; v < 8; ++v) {
            float y = fmaxf(acc[nt][v], 0.0f);
            Out[(size_t)(row0 + half * 8 + v) * DD + n] = y;
        }
    }
}

extern "C" void kernel_launch(void* const* d_in, const int* in_sizes, int n_in,
                              void* d_out, int out_size, void* d_ws, size_t ws_size,
                              hipStream_t stream) {
    const float* x   = (const float*)d_in[0];
    const int*   ei  = (const int*)d_in[1];
    const float* W1  = (const float*)d_in[2];
    const float* b1  = (const float*)d_in[3];
    const float* gnw = (const float*)d_in[4];
    const float* gnb = (const float*)d_in[5];
    const float* gna = (const float*)d_in[6];
    const float* W2  = (const float*)d_in[7];
    const float* b2  = (const float*)d_in[8];
    float* out = (float*)d_out;

    char* ws = (char*)d_ws;
    const size_t ndf = (size_t)NN * DD * sizeof(float);               // 51.2 MB
    size_t o = 0;
    float*  acc  = (float*)(ws + o);  o += ndf;                       // aggregation / gemm1 buffer
    float*  hbuf = (float*)(ws + o);  o += ndf;                       // ping-pong activations
    __bf16* w1t  = (__bf16*)(ws + o); o += (size_t)LL * DD * DD * 2;
    __bf16* w2t  = (__bf16*)(ws + o); o += (size_t)LL * DD * DD * 2;
    float*  stats = (float*)(ws + o); o += 4 * DD * sizeof(float);
    unsigned* deg    = (unsigned*)(ws + o); o += (size_t)NN * 4;
    unsigned* off    = (unsigned*)(ws + o); o += (size_t)NN * 4;
    unsigned* cursor = (unsigned*)(ws + o); o += (size_t)NN * 4;
    int*      srcs   = (int*)(ws + o);      o += (size_t)EE * 4;
    float* colsum   = stats;
    float* colsumsq = stats + DD;
    float* scl      = stats + 2 * DD;
    float* shf      = stats + 3 * DD;

    // ---- once per call: weight convert + CSR build of in-edges by dst ----
    k_convert<<<(LL * DD * DD) / 256, 256, 0, stream>>>(W1, W2, w1t, w2t);
    k_zero_deg<<<(NN + 255) / 256, 256, 0, stream>>>(deg);
    k_hist<<<(EE + 255) / 256, 256, 0, stream>>>(ei, deg);
    k_scan<<<1, 1024, 0, stream>>>(deg, off, cursor);
    k_fill<<<(EE + 255) / 256, 256, 0, stream>>>(ei, cursor, srcs);

    const int gemm_blocks = (NTILES + 7) / 8;     // 782 blocks x 8 waves
    const float* hin = x;
    for (int l = 0; l < LL; ++l) {
        // h1->hbuf, h2->out, h3->hbuf, h4->out, h5->out (in==out on last layer is
        // safe: aggregate reads hin before gemm2 overwrites it, stream-ordered)
        float* hout = (l == LL - 1) ? out : ((l & 1) ? out : hbuf);

        k_aggregate<<<NN / 8, 256, 0, stream>>>(hin, off, deg, srcs, acc);
        k_zero_stats<<<1, 256, 0, stream>>>(stats);
        k_gemm1<<<gemm_blocks, 256, 0, stream>>>(acc, w1t + (size_t)l * DD * DD,
                                                 b1 + l * DD, acc, colsum, colsumsq);
        k_finalize<<<1, DD, 0, stream>>>(colsum, colsumsq, gnw + l * DD, gnb + l * DD,
                                         gna + l * DD, scl, shf);
        k_gemm2<<<gemm_blocks, 256, 0, stream>>>(acc, w2t + (size_t)l * DD * DD,
                                                 b2 + l * DD, scl, shf, hout);
        hin = hout;
    }
}
